// Transolver_block_3401614098976
// MI455X (gfx1250) — compile-verified, matrix operand order checked by reading, not run on hardware
//
#include <hip/hip_runtime.h>
#include <hip/hip_bf16.h>
#include <math.h>

#define TOK_TOTAL (4 * 65536)   // B*N
#define C_ 256
#define H_ 8
#define D_ 32
#define G_ 32
#define F_ 1024
#define EPS_ 1e-5f

typedef __attribute__((ext_vector_type(16))) __bf16 v16bf;
typedef __attribute__((ext_vector_type(8)))  __bf16 v8bf;
typedef __attribute__((ext_vector_type(8)))  float  v8f;
typedef __attribute__((ext_vector_type(4)))  int    v4i;

#define DEV static __device__ __forceinline__

// ---- gfx1250 async global->LDS path (guarded; falls back to sync copy) ----
#if defined(__has_builtin)
#  if __has_builtin(__builtin_amdgcn_global_load_async_to_lds_b128) && \
      __has_builtin(__builtin_amdgcn_s_wait_asynccnt)
#    define HAVE_ASYNC_LDS 1
#  endif
#endif
#ifndef HAVE_ASYNC_LDS
#  define HAVE_ASYNC_LDS 0
#endif

#if HAVE_ASYNC_LDS
typedef __attribute__((address_space(1))) v4i g_v4i;   // global int4
typedef __attribute__((address_space(3))) v4i l_v4i;   // LDS int4
#endif

// Async-copy one 16-byte chunk from global to LDS (ASYNCcnt tracked), or
// plain vector copy when the builtin is unavailable.
DEV void copy16_g2l(__bf16* lds_dst, const __bf16* gsrc) {
#if HAVE_ASYNC_LDS
  __builtin_amdgcn_global_load_async_to_lds_b128(
      (g_v4i*)gsrc, (l_v4i*)lds_dst, 0, 0);
#else
  *(v8bf*)lds_dst = *(const v8bf*)gsrc;
#endif
}

DEV void async_copy_fence() {
#if HAVE_ASYNC_LDS
  __builtin_amdgcn_s_wait_asynccnt(0);
#endif
}

DEV v8f vzero8() {
  v8f z;
#pragma unroll
  for (int i = 0; i < 8; i++) z[i] = 0.f;
  return z;
}

// 16-bit A/B fragment, contiguous-K source: lane holds row/col `row` of the
// matrix; element e -> k = k0 + half*8 + e (e<8) or k0 + 16 + half*8 + (e-8).
DEV v16bf frag_ld(const __bf16* rowp, int k0, int half) {
  const __bf16* p = rowp + k0 + half * 8;
  v8bf lo = *(const v8bf*)p;
  v8bf hi = *(const v8bf*)(p + 16);
  v16bf f;
#pragma unroll
  for (int i = 0; i < 8; i++) { f[i] = lo[i]; f[8 + i] = hi[i]; }
  return f;
}

// Same fragment but K runs with a stride (gather), k0 == 0.
DEV v16bf frag_gather(const __bf16* base, int stride, int half) {
  v16bf f;
#pragma unroll
  for (int e = 0; e < 16; e++) {
    int k = (e < 8) ? (half * 8 + e) : (16 + half * 8 + (e - 8));
    f[e] = base[k * stride];
  }
  return f;
}

DEV v8f wmma_bf(v16bf a, v16bf b, v8f c) {
  return __builtin_amdgcn_wmma_f32_16x16x32_bf16(false, a, false, b, (short)0, c,
                                                 false, false);
}

// out[MT*16 x 32] tile GEMM: A (bf16, lda) x W^T (W bf16 [outRows][ldb]).
// Wave covers output cols n0..n0+31. acc laid out [mi][ni].
template <int MT>
DEV void gemm_nk(const __bf16* A, int lda, const __bf16* Wb, int ldb, int K,
                 int n0, int lane, v8f (&acc)[MT][2]) {
  int half = lane >> 4, l16 = lane & 15;
  for (int k0 = 0; k0 < K; k0 += 32) {
    v16bf b0 = frag_ld(Wb + (size_t)(n0 + l16) * ldb, k0, half);
    v16bf b1 = frag_ld(Wb + (size_t)(n0 + 16 + l16) * ldb, k0, half);
#pragma unroll
    for (int mi = 0; mi < MT; mi++) {
      v16bf a = frag_ld(A + (size_t)(mi * 16 + l16) * lda, k0, half);
      acc[mi][0] = wmma_bf(a, b0, acc[mi][0]);
      acc[mi][1] = wmma_bf(a, b1, acc[mi][1]);
    }
  }
}

// One wave LayerNorms one row of 256 f32 -> bf16 (wave32 shuffle reduction).
DEV void ln_row_256(const float* src, const float* g, const float* b,
                    __bf16* dst, int lane) {
  float v[8];
  float s = 0.f;
#pragma unroll
  for (int i = 0; i < 8; i++) { v[i] = src[lane + 32 * i]; s += v[i]; }
#pragma unroll
  for (int off = 16; off; off >>= 1) s += __shfl_xor(s, off, 32);
  float mean = s * (1.f / 256.f);
  float q = 0.f;
#pragma unroll
  for (int i = 0; i < 8; i++) { float d = v[i] - mean; q += d * d; }
#pragma unroll
  for (int off = 16; off; off >>= 1) q += __shfl_xor(q, off, 32);
  float rinv = rsqrtf(q * (1.f / 256.f) + EPS_);
#pragma unroll
  for (int i = 0; i < 8; i++) {
    int c = lane + 32 * i;
    dst[c] = (__bf16)(((v[i] - mean) * rinv) * g[c] + b[c]);
  }
}

// ---------------- K0: weight f32->bf16 + accumulator zeroing ----------------
__global__ __launch_bounds__(256) void k0_prep(
    const float* __restrict__ Wx, const float* __restrict__ Ws,
    const float* __restrict__ Wo, const float* __restrict__ W1,
    const float* __restrict__ W2, __bf16* __restrict__ Wxb,
    __bf16* __restrict__ Wsb, __bf16* __restrict__ Wob,
    __bf16* __restrict__ W1b, __bf16* __restrict__ W2b,
    float* __restrict__ stn, float* __restrict__ sn) {
  int idx = blockIdx.x * 256 + threadIdx.x;   // grid sized to 262144 threads
  W1b[idx] = (__bf16)W1[idx];
  W2b[idx] = (__bf16)W2[idx];
  if (idx < 65536) { Wxb[idx] = (__bf16)Wx[idx]; Wob[idx] = (__bf16)Wo[idx]; }
  if (idx < 32768) stn[idx] = 0.f;
  if (idx < 1024)  { Wsb[idx] = (__bf16)Ws[idx]; sn[idx] = 0.f; }
}

// ---------------- K1: LN1 + xm = ln(fx) @ Wx^T + bx  (bf16 out) -------------
__global__ __launch_bounds__(256) void k1_ln_proj(
    const float* __restrict__ fx, const float* __restrict__ g1,
    const float* __restrict__ b1, const __bf16* __restrict__ Wxb,
    const float* __restrict__ bx, __bf16* __restrict__ xm) {
  __shared__ alignas(16) __bf16 x1s[32][264];
  int tid = threadIdx.x, lane = tid & 31, wave = tid >> 5;
  int half = lane >> 4, l16 = lane & 15;
  size_t tok0 = (size_t)blockIdx.x * 32;
  __builtin_prefetch(Wxb + (size_t)wave * 32 * 256, 0, 0);
#pragma unroll
  for (int rr = 0; rr < 4; rr++) {
    int row = wave * 4 + rr;
    ln_row_256(fx + (tok0 + row) * C_, g1, b1, &x1s[row][0], lane);
  }
  __syncthreads();
  v8f acc[2][2];
#pragma unroll
  for (int mi = 0; mi < 2; mi++) { acc[mi][0] = vzero8(); acc[mi][1] = vzero8(); }
  gemm_nk<2>(&x1s[0][0], 264, Wxb, 256, 256, wave * 32, lane, acc);
#pragma unroll
  for (int mi = 0; mi < 2; mi++)
#pragma unroll
    for (int ni = 0; ni < 2; ni++) {
      int col = wave * 32 + ni * 16 + l16;
      float bxv = bx[col];
#pragma unroll
      for (int r = 0; r < 8; r++) {
        int row = mi * 16 + half * 8 + r;
        xm[(tok0 + row) * C_ + col] = (__bf16)(acc[mi][ni][r] + bxv);
      }
    }
}

// ------- K2: slice softmax + atomic accumulation of st_num / sn -------------
__global__ __launch_bounds__(256) void k2_slice(
    const __bf16* __restrict__ xm, const __bf16* __restrict__ Wsb,
    const float* __restrict__ bs, const float* __restrict__ temp,
    float* __restrict__ stn, float* __restrict__ sn) {
  __shared__ alignas(16) __bf16 xms[32][264];   // rows 16..31 zero pad (K pad)
  __shared__ float lgs[8][16][33];
  __shared__ alignas(16) __bf16 sws[8][32][40]; // rows 16..31 zero pad
  int tid = threadIdx.x, lane = tid & 31, wave = tid >> 5;
  int half = lane >> 4, l16 = lane & 15;
  size_t tok0 = (size_t)blockIdx.x * 16;
  int b = (int)(tok0 >> 16);
  const __bf16* src = xm + tok0 * C_;
  for (int i = tid; i < 16 * 32; i += 256) {        // 16 rows x 32 chunks of 8
    int r = i >> 5, c = (i & 31) * 8;
    copy16_g2l(&xms[r][c], src + r * C_ + c);       // async -> LDS (ASYNCcnt)
  }
  for (int i = tid; i < 16 * 256; i += 256) xms[16 + (i >> 8)][i & 255] = (__bf16)0.f;
  for (int i = tid; i < 8 * 16 * 40; i += 256) {
    int h = i / 640, r = i % 640;
    sws[h][16 + r / 40][r % 40] = (__bf16)0.f;
  }
  async_copy_fence();
  __syncthreads();
  int h = wave;  // one head per wave
  // logits[16 tok x 32 g] = xm_h @ Ws^T  (K=32, one WMMA per 16x16 tile)
  v16bf a  = frag_ld(&xms[l16][h * 32], 0, half);
  v16bf w0 = frag_ld(Wsb + l16 * 32, 0, half);
  v16bf w1 = frag_ld(Wsb + (16 + l16) * 32, 0, half);
  v8f lz = vzero8();
  v8f l0 = wmma_bf(a, w0, lz);
  v8f l1 = wmma_bf(a, w1, lz);
#pragma unroll
  for (int r = 0; r < 8; r++) {
    lgs[h][half * 8 + r][l16]      = l0[r];
    lgs[h][half * 8 + r][16 + l16] = l1[r];
  }
  float invt = 1.f / temp[h];
  if (lane < 16) {
    int t = lane;
    float rowv[32];
    float mx = -1e30f;
#pragma unroll
    for (int g = 0; g < 32; g++) {
      float x = (lgs[h][t][g] + bs[g]) * invt;
      rowv[g] = x; mx = fmaxf(mx, x);
    }
    float s = 0.f;
#pragma unroll
    for (int g = 0; g < 32; g++) { float e = __expf(rowv[g] - mx); rowv[g] = e; s += e; }
    float rs = 1.f / s;
#pragma unroll
    for (int g = 0; g < 32; g++) {
      float w = rowv[g] * rs;
      lgs[h][t][g] = w;
      sws[h][t][g] = (__bf16)w;
    }
  }
  {
    float s = 0.f;  // lane == g: column sum for sn
#pragma unroll
    for (int t = 0; t < 16; t++) s += lgs[h][t][lane];
    atomicAdd(&sn[(b * H_ + h) * G_ + lane], s);
  }
  // st_num[g][d] += sw^T @ xm_h  (K = 32 tokens, zero-padded)
  v8f acc[2][2];
#pragma unroll
  for (int mi = 0; mi < 2; mi++) { acc[mi][0] = vzero8(); acc[mi][1] = vzero8(); }
  v16bf a0 = frag_gather(&sws[h][0][l16], 40, half);
  v16bf a1 = frag_gather(&sws[h][0][16 + l16], 40, half);
  v16bf b0 = frag_gather(&xms[0][h * 32 + l16], 264, half);
  v16bf b1 = frag_gather(&xms[0][h * 32 + 16 + l16], 264, half);
  acc[0][0] = wmma_bf(a0, b0, acc[0][0]);
  acc[0][1] = wmma_bf(a0, b1, acc[0][1]);
  acc[1][0] = wmma_bf(a1, b0, acc[1][0]);
  acc[1][1] = wmma_bf(a1, b1, acc[1][1]);
#pragma unroll
  for (int mi = 0; mi < 2; mi++)
#pragma unroll
    for (int ni = 0; ni < 2; ni++)
#pragma unroll
      for (int r = 0; r < 8; r++) {
        int g = mi * 16 + half * 8 + r;
        int d = ni * 16 + l16;
        atomicAdd(&stn[((b * H_ + h) * G_ + g) * D_ + d], acc[mi][ni][r]);
      }
}

// ------- K3: tiny 32x32 attention over slice tokens, per (b,h) --------------
__global__ __launch_bounds__(32) void k3_attn(
    const float* __restrict__ stn, const float* __restrict__ sn,
    const float* __restrict__ Wq, const float* __restrict__ Wk,
    const float* __restrict__ Wv, __bf16* __restrict__ ost) {
  __shared__ float st[32][33], qs[32][33], ks[32][33], vs[32][33];
  int bh = blockIdx.x, g = threadIdx.x;
  float rs = 1.f / (sn[bh * G_ + g] + EPS_);
  for (int d = 0; d < 32; d++) st[g][d] = stn[(bh * G_ + g) * D_ + d] * rs;
  __syncthreads();
  for (int e = 0; e < 32; e++) {
    float aq = 0.f, ak = 0.f, av = 0.f;
    for (int d = 0; d < 32; d++) {
      float x = st[g][d];
      aq += x * Wq[e * 32 + d];
      ak += x * Wk[e * 32 + d];
      av += x * Wv[e * 32 + d];
    }
    qs[g][e] = aq; ks[g][e] = ak; vs[g][e] = av;
  }
  __syncthreads();
  const float scale = 0.17677669529663687f;  // 32^-0.5
  float sc[32];
  float mx = -1e30f;
  for (int f = 0; f < 32; f++) {
    float s = 0.f;
    for (int e = 0; e < 32; e++) s += qs[g][e] * ks[f][e];
    s *= scale; sc[f] = s; mx = fmaxf(mx, s);
  }
  float ssum = 0.f;
  for (int f = 0; f < 32; f++) { float e = __expf(sc[f] - mx); sc[f] = e; ssum += e; }
  float inv = 1.f / ssum;
  for (int d = 0; d < 32; d++) {
    float o = 0.f;
    for (int f = 0; f < 32; f++) o += sc[f] * vs[f][d];
    ost[(bh * G_ + g) * D_ + d] = (__bf16)(o * inv);
  }
}

// ------- K5: recompute sw, ox = sw @ out_st, then @Wo^T + bo + fx -----------
__global__ __launch_bounds__(256) void k5_broadcast_wo(
    const __bf16* __restrict__ xm, const __bf16* __restrict__ Wsb,
    const float* __restrict__ bs, const float* __restrict__ temp,
    const __bf16* __restrict__ ost, const __bf16* __restrict__ Wob,
    const float* __restrict__ bo, const float* __restrict__ fx,
    float* __restrict__ fx2) {
  __shared__ alignas(16) __bf16 xms[16][264];
  __shared__ alignas(16) __bf16 oxs[16][264];
  __shared__ float lgs[8][16][33];
  __shared__ alignas(16) __bf16 sws[8][16][40];
  __shared__ alignas(16) __bf16 ostT[8][32][32];  // [h][d][g]
  int tid = threadIdx.x, lane = tid & 31, wave = tid >> 5;
  int half = lane >> 4, l16 = lane & 15;
  size_t tok0 = (size_t)blockIdx.x * 16;
  int b = (int)(tok0 >> 16);
  const __bf16* src = xm + tok0 * C_;
  for (int i = tid; i < 16 * 32; i += 256) {
    int r = i >> 5, c = (i & 31) * 8;
    copy16_g2l(&xms[r][c], src + r * C_ + c);     // async -> LDS (ASYNCcnt)
  }
  for (int i = tid; i < 8 * 32 * 32; i += 256) {
    int h = i >> 10, g = (i >> 5) & 31, d = i & 31;
    ostT[h][d][g] = ost[((size_t)(b * H_ + h) * G_ + g) * D_ + d];
  }
  async_copy_fence();
  __syncthreads();
  int h = wave;
  v16bf a  = frag_ld(&xms[l16][h * 32], 0, half);
  v16bf w0 = frag_ld(Wsb + l16 * 32, 0, half);
  v16bf w1 = frag_ld(Wsb + (16 + l16) * 32, 0, half);
  v8f lz = vzero8();
  v8f l0 = wmma_bf(a, w0, lz);
  v8f l1 = wmma_bf(a, w1, lz);
#pragma unroll
  for (int r = 0; r < 8; r++) {
    lgs[h][half * 8 + r][l16]      = l0[r];
    lgs[h][half * 8 + r][16 + l16] = l1[r];
  }
  float invt = 1.f / temp[h];
  if (lane < 16) {
    int t = lane;
    float rowv[32];
    float mx = -1e30f;
#pragma unroll
    for (int g = 0; g < 32; g++) {
      float x = (lgs[h][t][g] + bs[g]) * invt;
      rowv[g] = x; mx = fmaxf(mx, x);
    }
    float s = 0.f;
#pragma unroll
    for (int g = 0; g < 32; g++) { float e = __expf(rowv[g] - mx); rowv[g] = e; s += e; }
    float rs = 1.f / s;
#pragma unroll
    for (int g = 0; g < 32; g++) sws[h][t][g] = (__bf16)(rowv[g] * rs);
  }
  // ox_h[16 tok x 32 d] = sw @ out_st_h  (K = G = 32)
  v16bf as  = frag_ld(&sws[h][l16][0], 0, half);
  v16bf ob0 = frag_ld(&ostT[h][l16][0], 0, half);
  v16bf ob1 = frag_ld(&ostT[h][16 + l16][0], 0, half);
  v8f o0 = wmma_bf(as, ob0, lz);
  v8f o1 = wmma_bf(as, ob1, lz);
#pragma unroll
  for (int r = 0; r < 8; r++) {
    int row = half * 8 + r;
    oxs[row][h * 32 + l16]      = (__bf16)o0[r];
    oxs[row][h * 32 + 16 + l16] = (__bf16)o1[r];
  }
  __syncthreads();
  v8f acc[1][2];
  acc[0][0] = vzero8(); acc[0][1] = vzero8();
  gemm_nk<1>(&oxs[0][0], 264, Wob, 256, 256, wave * 32, lane, acc);
#pragma unroll
  for (int ni = 0; ni < 2; ni++) {
    int col = wave * 32 + ni * 16 + l16;
    float bov = bo[col];
#pragma unroll
    for (int r = 0; r < 8; r++) {
      int row = half * 8 + r;
      size_t idx = (tok0 + row) * C_ + col;
      fx2[idx] = acc[0][ni][r] + bov + fx[idx];
    }
  }
}

// ------- K6: LN2 + MLP (W1/GELU/W2) + residual ------------------------------
__global__ __launch_bounds__(256) void k6_mlp(
    const float* __restrict__ fx2, const float* __restrict__ g2,
    const float* __restrict__ b2ln, const __bf16* __restrict__ W1b,
    const float* __restrict__ b1, const __bf16* __restrict__ W2b,
    const float* __restrict__ b2, float* __restrict__ out) {
  __shared__ alignas(16) __bf16 x2s[16][264];
  __shared__ alignas(16) __bf16 hs[16][1056];
  int tid = threadIdx.x, lane = tid & 31, wave = tid >> 5;
  int half = lane >> 4, l16 = lane & 15;
  size_t tok0 = (size_t)blockIdx.x * 16;
  __builtin_prefetch(W2b + (size_t)wave * 32 * 1024, 0, 0);
#pragma unroll
  for (int rr = 0; rr < 2; rr++) {
    int row = wave * 2 + rr;
    ln_row_256(fx2 + (tok0 + row) * C_, g2, b2ln, &x2s[row][0], lane);
  }
  __syncthreads();
  for (int ch = 0; ch < 4; ch++) {   // 4 chunks of 256 hidden cols
    v8f acc[1][2];
    acc[0][0] = vzero8(); acc[0][1] = vzero8();
    gemm_nk<1>(&x2s[0][0], 264, W1b + (size_t)(ch * 256) * 256, 256, 256,
               wave * 32, lane, acc);
#pragma unroll
    for (int ni = 0; ni < 2; ni++) {
      int col = ch * 256 + wave * 32 + ni * 16 + l16;
      float bb = b1[col];
#pragma unroll
      for (int r = 0; r < 8; r++) {
        int row = half * 8 + r;
        float x = acc[0][ni][r] + bb;
        float ge = 0.5f * x * (1.f + erff(x * 0.70710678118654752f));
        hs[row][col] = (__bf16)ge;
      }
    }
  }
  __syncthreads();
  v8f acc[1][2];
  acc[0][0] = vzero8(); acc[0][1] = vzero8();
  gemm_nk<1>(&hs[0][0], 1056, W2b, 1024, 1024, wave * 32, lane, acc);
#pragma unroll
  for (int ni = 0; ni < 2; ni++) {
    int col = wave * 32 + ni * 16 + l16;
    float bb = b2[col];
#pragma unroll
    for (int r = 0; r < 8; r++) {
      int row = half * 8 + r;
      size_t idx = (tok0 + row) * C_ + col;
      out[idx] = acc[0][ni][r] + bb + fx2[idx];
    }
  }
}

// ---------------------------------------------------------------------------
extern "C" void kernel_launch(void* const* d_in, const int* in_sizes, int n_in,
                              void* d_out, int out_size, void* d_ws, size_t ws_size,
                              hipStream_t stream) {
  (void)in_sizes; (void)n_in; (void)out_size; (void)ws_size;
  const float* fx   = (const float*)d_in[0];
  const float* g1   = (const float*)d_in[1];
  const float* b1ln = (const float*)d_in[2];
  const float* Wx   = (const float*)d_in[3];
  const float* bx   = (const float*)d_in[4];
  const float* Ws   = (const float*)d_in[5];
  const float* bs   = (const float*)d_in[6];
  const float* temp = (const float*)d_in[7];
  const float* Wq   = (const float*)d_in[8];
  const float* Wk   = (const float*)d_in[9];
  const float* Wv   = (const float*)d_in[10];
  const float* Wo   = (const float*)d_in[11];
  const float* bo   = (const float*)d_in[12];
  const float* g2   = (const float*)d_in[13];
  const float* b2ln = (const float*)d_in[14];
  const float* W1   = (const float*)d_in[15];
  const float* b1   = (const float*)d_in[16];
  const float* W2   = (const float*)d_in[17];
  const float* b2   = (const float*)d_in[18];

  char* ws = (char*)d_ws;
  size_t off = 0;
  auto carve = [&](size_t bytes) -> void* {
    void* p = ws + off;
    off += (bytes + 255) & ~(size_t)255;
    return p;
  };
  __bf16* Wxb  = (__bf16*)carve(65536 * 2);
  __bf16* Wob  = (__bf16*)carve(65536 * 2);
  __bf16* W1b  = (__bf16*)carve(262144 * 2);
  __bf16* W2b  = (__bf16*)carve(262144 * 2);
  __bf16* Wsb  = (__bf16*)carve(1024 * 2);
  float*  stn  = (float*)carve(32768 * 4);
  float*  snb  = (float*)carve(1024 * 4);
  __bf16* ostb = (__bf16*)carve(8192 * 2);
  __bf16* xmb  = (__bf16*)carve((size_t)TOK_TOTAL * 256 * 2);
  float*  fx2  = (float*)carve((size_t)TOK_TOTAL * 256 * 4);

  k0_prep<<<1024, 256, 0, stream>>>(Wx, Ws, Wo, W1, W2, Wxb, Wsb, Wob, W1b, W2b,
                                    stn, snb);
  k1_ln_proj<<<TOK_TOTAL / 32, 256, 0, stream>>>(fx, g1, b1ln, Wxb, bx, xmb);
  k2_slice<<<TOK_TOTAL / 16, 256, 0, stream>>>(xmb, Wsb, bs, temp, stn, snb);
  k3_attn<<<32, 32, 0, stream>>>(stn, snb, Wq, Wk, Wv, ostb);
  k5_broadcast_wo<<<TOK_TOTAL / 16, 256, 0, stream>>>(xmb, Wsb, bs, temp, ostb,
                                                      Wob, bo, fx, fx2);
  k6_mlp<<<TOK_TOTAL / 16, 256, 0, stream>>>(fx2, g2, b2ln, W1b, b1, W2b, b2,
                                             (float*)d_out);
}
